// TPoseHuman_68324339745351
// MI455X (gfx1250) — matrix operand order repaired and checked by reading, not compiled, not run on hardware
//
#include <hip/hip_runtime.h>

typedef __attribute__((ext_vector_type(16))) _Float16 v16h;
typedef __attribute__((ext_vector_type(8)))  _Float16 v8h;
typedef __attribute__((ext_vector_type(4)))  _Float16 v4h;
typedef __attribute__((ext_vector_type(8)))  float    v8f;

#define PPARTS 6
#define DDIM   256
#define RAWD   20
#define OUTC   27        // 20 raw + 1 occ + 6 tocc
#define PART_H 81920     // halves per part in ws: W1 8192 + W2 8*8192 + W3occ 8*1024
#define W2_H   8192      // halves, one 32x256 chunk
#define W3_H   1024      // halves, one 32x32 chunk

// D = A(16x32 f16) * B(32x16 f16) + C(16x16 f32)
static __device__ inline v8f wmma16(v16h a, v16h b, v8f c) {
  return __builtin_amdgcn_wmma_f32_16x16x32_f16(
      /*neg_a=*/false, a, /*neg_b=*/false, b,
      /*c_mod=*/(short)0, c, /*reuse_a=*/false, /*reuse_b=*/false);
}

// A fragment from a row-major [16][256] f16 tile: two contiguous 8-half runs.
static __device__ inline v16h load_a_frag(const _Float16* hw, int lane, int k0) {
  const int g = lane >> 4, m = lane & 15;
  const _Float16* base = hw + m * DDIM + k0 + (g << 4);
  v8h lo = *(const v8h*)(base);
  v8h hi = *(const v8h*)(base + 8);
  return __builtin_shufflevector(lo, hi, 0, 1, 2, 3, 4, 5, 6, 7,
                                 8, 9, 10, 11, 12, 13, 14, 15);
}

// B fragment from a pre-swizzled tile: one contiguous 32-byte run per (jn, lane).
static __device__ inline v16h load_b_frag_swz(const _Float16* Wt, int jn, int lane) {
  return *(const v16h*)(Wt + (((jn << 5) + lane) << 4));
}

// One async 512-byte wave copy: each lane moves 16 bytes global -> LDS.
// ASYNCcnt-tracked (CDNA5 §10.7 / 15.18 GLOBAL_LOAD_ASYNC_TO_LDS_B128).
static __device__ inline void async_ld_512(unsigned lds_off, unsigned long long gaddr) {
  asm volatile("global_load_async_to_lds_b128 %0, %1, off"
               :: "v"(lds_off), "v"(gaddr) : "memory");
}
#define WAIT_ASYNC_0() asm volatile("s_wait_asynccnt 0x0" ::: "memory")
#define WAIT_ASYNC_1() asm volatile("s_wait_asynccnt 0x1" ::: "memory")
#define WAIT_ASYNC_4() asm volatile("s_wait_asynccnt 0x4" ::: "memory")

// ---------------------------------------------------------------------------
// One-shot: convert all weights to f16 in B-fragment-swizzled layout in d_ws.
// Grid: 6 parts x 10 block-types (0: W1, 1..8: W2 chunk, 9: W3+Wocc).
// ---------------------------------------------------------------------------
__global__ __launch_bounds__(256) void convert_weights_kernel(
    const float* __restrict__ W1, const float* __restrict__ W2,
    const float* __restrict__ W3, const float* __restrict__ Wocc,
    _Float16* __restrict__ ws)
{
  const int b = blockIdx.x;
  const int p = b / 10, t = b % 10;
  const int tid = threadIdx.x;
  const int jn = tid >> 4, c15 = tid & 15;
  _Float16* pbase = ws + (long)p * PART_H;

  if (t == 0) {                       // W1: 17 rows padded to 32
    const float* W1p = W1 + (long)p * 17 * DDIM;
    v16h h0, h1;
#pragma unroll
    for (int i = 0; i < 16; ++i) h0[i] = (_Float16)W1p[i * DDIM + tid];
    h1 = (v16h)(_Float16)0.f;
    h1[0] = (_Float16)W1p[16 * DDIM + tid];
    *(v16h*)(pbase + (((jn << 5) + c15) << 4)) = h0;
    *(v16h*)(pbase + (((jn << 5) + c15 + 16) << 4)) = h1;
  } else if (t <= 8) {                // W2 chunk kc = t-1 (32 x 256)
    const int kc = t - 1;
    const float* src = W2 + (long)p * DDIM * DDIM + (kc * 32) * DDIM + tid;
    v16h h0, h1;
#pragma unroll
    for (int i = 0; i < 16; ++i) h0[i] = (_Float16)src[i * DDIM];
#pragma unroll
    for (int i = 0; i < 16; ++i) h1[i] = (_Float16)src[(16 + i) * DDIM];
    _Float16* dst = pbase + W2_H + (long)kc * W2_H;
    *(v16h*)(dst + (((jn << 5) + c15) << 4)) = h0;
    *(v16h*)(dst + (((jn << 5) + c15 + 16) << 4)) = h1;
  } else {                            // W3 (20 cols) + Wocc (col 20), pad to 32
    const int scol = tid >> 3;        // 0..31
    const int skl  = (tid & 7) << 2;  // 0,4,...,28
    const float* W3p   = W3 + (long)p * DDIM * RAWD;
    const float* Woccp = Wocc + (long)p * DDIM;
    const long swz = ((((scol >> 4) << 5) + (scol & 15) + ((skl >> 4) << 4)) << 4) + (skl & 15);
    for (int kc = 0; kc < 8; ++kc) {
      v4h hv;
#pragma unroll
      for (int i = 0; i < 4; ++i) {
        const int krow = kc * 32 + skl + i;
        float v = 0.f;
        if (scol < RAWD)       v = W3p[krow * RAWD + scol];
        else if (scol == RAWD) v = Woccp[krow];
        hv[i] = (_Float16)v;
      }
      *(v4h*)(pbase + 9 * W2_H + (long)kc * W3_H + swz) = hv;
    }
  }
}

// ---------------------------------------------------------------------------
// Main fused MLP kernel. USE_WS: weights come pre-swizzled f16 from d_ws and
// are staged with double-buffered GLOBAL_LOAD_ASYNC_TO_LDS_B128.
// ---------------------------------------------------------------------------
template <bool USE_WS>
__global__ __launch_bounds__(256, 1) void tpose_mlp_wmma_kernel(
    const float* __restrict__ tpts, const float* __restrict__ bigpts,
    const float* __restrict__ viewdir, const float* __restrict__ frame,
    const float* __restrict__ W1, const float* __restrict__ b1,
    const float* __restrict__ W2, const float* __restrict__ b2,
    const float* __restrict__ W3, const float* __restrict__ b3,
    const float* __restrict__ Wocc, const float* __restrict__ bocc,
    const int* __restrict__ tflag, const _Float16* __restrict__ wsw,
    float* __restrict__ out, int N)
{
  extern __shared__ char smem[];
  _Float16* Wt   = (_Float16*)smem;                          // 2 x 8192 halves (32 KB), double buffer
  _Float16* hb   = (_Float16*)(smem + 2 * W2_H * 2);         // 8 waves x 16 x 256 f16 (64 KB)
  float*    occs = (float*)(smem + 2 * W2_H * 2 + 8 * 16 * DDIM * 2); // 8x16x6 f32

  const int tid  = threadIdx.x;
  const int wave = tid >> 5, lane = tid & 31;
  const int g = lane >> 4, ln = lane & 15;
  _Float16* hw = hb + wave * 16 * DDIM;
  const int wrow0 = blockIdx.x * 128 + wave * 16;
  const unsigned ldsWt = (unsigned)(size_t)Wt;               // LDS byte offset of Wt

  // fallback-path staging pointers (column `tid` of a 32xK chunk)
  const int sjn = tid >> 4, sc15 = tid & 15;
  _Float16* sdst0 = Wt + (((sjn << 5) + sc15) << 4);
  _Float16* sdst1 = Wt + (((sjn << 5) + sc15 + 16) << 4);

  v8f accOut0 = {};
  v8f accOut1 = {};
  const v8f vzero = {};

  for (int p = 0; p < PPARTS; ++p) {
    const _Float16* wsp = USE_WS ? (wsw + (size_t)p * PART_H) : (const _Float16*)0;

    if (USE_WS) {
      // issue W1 tile -> buf0 and W2 chunk0 -> buf1 (4 x 512B ops each per wave)
      const unsigned l0 = ldsWt + wave * 2048 + lane * 16;
      const unsigned long long gW1 = (unsigned long long)(size_t)wsp + wave * 2048 + lane * 16;
#pragma unroll
      for (int i = 0; i < 4; ++i) async_ld_512(l0 + i * 512, gW1 + i * 512);
      const unsigned l1 = ldsWt + 2 * W2_H + wave * 2048 + lane * 16;
      const unsigned long long gC0 = (unsigned long long)(size_t)(wsp + W2_H) + wave * 2048 + lane * 16;
#pragma unroll
      for (int i = 0; i < 4; ++i) async_ld_512(l1 + i * 512, gC0 + i * 512);
    }

    // ---------- stage input X tile (16 x 32, cols 17..31 = 0) into hw ----------
    if (lane < 16) {
      const int n = wrow0 + ln;
      _Float16* xr = hw + ln * DDIM;
      if (n < N) {
        const long base = ((long)n * PPARTS + p) * 3;
        xr[0] = (_Float16)tpts[base + 0];
        xr[1] = (_Float16)tpts[base + 1];
        xr[2] = (_Float16)tpts[base + 2];
#pragma unroll
        for (int c = 0; c < 8; ++c) xr[3 + c] = (_Float16)frame[c];
        xr[11] = (_Float16)bigpts[base + 0];
        xr[12] = (_Float16)bigpts[base + 1];
        xr[13] = (_Float16)bigpts[base + 2];
        xr[14] = (_Float16)viewdir[base + 0];
        xr[15] = (_Float16)viewdir[base + 1];
        xr[16] = (_Float16)viewdir[base + 2];
      } else {
#pragma unroll
        for (int c = 0; c < 17; ++c) xr[c] = (_Float16)0.f;
      }
    } else {
      _Float16* xr = hw + ln * DDIM;
#pragma unroll
      for (int c = 17; c < 32; ++c) xr[c] = (_Float16)0.f;
    }

    if (USE_WS) {
      WAIT_ASYNC_4();           // W1 tile landed (chunk0 may still be in flight)
    } else {
      const float* W1p = W1 + (long)p * 17 * DDIM;
      v16h h0, h1;
#pragma unroll
      for (int i = 0; i < 16; ++i) h0[i] = (_Float16)W1p[i * DDIM + tid];
      h1 = (v16h)(_Float16)0.f;
      h1[0] = (_Float16)W1p[16 * DDIM + tid];
      *(v16h*)sdst0 = h0;
      *(v16h*)sdst1 = h1;
    }
    __syncthreads();

    // ---------- layer 1: (16x32) x (32x256) ----------
    v8f acc[16];
#pragma unroll
    for (int jn = 0; jn < 16; ++jn) acc[jn] = vzero;
    {
      v16h a = load_a_frag(hw, lane, 0);
#pragma unroll
      for (int jn = 0; jn < 16; ++jn) {
        v16h b = load_b_frag_swz(Wt, jn, lane);
        acc[jn] = wmma16(a, b, acc[jn]);
      }
    }
    __syncthreads();   // all waves done with buf0 / X region
    {
      const float* b1p = b1 + (long)p * DDIM;
#pragma unroll
      for (int jn = 0; jn < 16; ++jn) {
        const int col = jn * 16 + ln;
        const float bias = b1p[col];
#pragma unroll
        for (int r = 0; r < 8; ++r) {
          const float h = acc[jn][r] + bias;
          hw[(r + g * 8) * DDIM + col] = (_Float16)fmaxf(h, 0.f);
        }
      }
    }

    // ---------- layer 2: (16x256) x (256x256), 8 chunks, double-buffered ----------
#pragma unroll
    for (int jn = 0; jn < 16; ++jn) acc[jn] = vzero;
    {
      const float* W2p = W2 + (long)p * DDIM * DDIM;
      for (int kc = 0; kc < 8; ++kc) {
        if (USE_WS) {
          // chunk kc lives in buf[(kc&1)^1]; prefetch chunk kc+1 into buf[kc&1]
          if (kc < 7) {
            const unsigned l = ldsWt + (unsigned)((kc & 1) * (2 * W2_H)) + wave * 2048 + lane * 16;
            const unsigned long long gsrc =
                (unsigned long long)(size_t)(wsp + W2_H + (size_t)(kc + 1) * W2_H) + wave * 2048 + lane * 16;
#pragma unroll
            for (int i = 0; i < 4; ++i) async_ld_512(l + i * 512, gsrc + i * 512);
            WAIT_ASYNC_4();
          } else {
            WAIT_ASYNC_0();
          }
          __syncthreads();
        } else {
          __syncthreads();
          {
            const float* src = W2p + (kc * 32) * DDIM + tid;
            v16h h0, h1;
#pragma unroll
            for (int i = 0; i < 16; ++i) h0[i] = (_Float16)src[i * DDIM];
#pragma unroll
            for (int i = 0; i < 16; ++i) h1[i] = (_Float16)src[(16 + i) * DDIM];
            *(v16h*)sdst0 = h0;
            *(v16h*)sdst1 = h1;
          }
          if (kc < 7) __builtin_prefetch(&W2p[(kc + 1) * 32 * DDIM + tid], 0, 1);
          __syncthreads();
        }
        const _Float16* curW = USE_WS ? (Wt + ((kc & 1) ^ 1) * W2_H) : Wt;
        v16h a = load_a_frag(hw, lane, kc * 32);
#pragma unroll
        for (int jn = 0; jn < 16; ++jn) {
          v16h b = load_b_frag_swz(curW, jn, lane);
          acc[jn] = wmma16(a, b, acc[jn]);
        }
        __syncthreads();   // buffer consumed; safe to overwrite next iteration
      }
    }
    {
      const float* b2p = b2 + (long)p * DDIM;
#pragma unroll
      for (int jn = 0; jn < 16; ++jn) {
        const int col = jn * 16 + ln;
        const float bias = b2p[col];
#pragma unroll
        for (int r = 0; r < 8; ++r) {
          const float h = acc[jn][r] + bias;
          hw[(r + g * 8) * DDIM + col] = (_Float16)fmaxf(h, 0.f);
        }
      }
    }

    // ---------- layer 3 + occ: (16x256) x (256x32), 8 chunks ----------
    v8f acc3[2];
    acc3[0] = vzero; acc3[1] = vzero;
    {
      const float* W3p   = W3 + (long)p * DDIM * RAWD;
      const float* Woccp = Wocc + (long)p * DDIM;
      const int scol = tid >> 3;
      const int skl  = (tid & 7) << 2;
      _Float16* sdst3 = Wt + ((((scol >> 4) << 5) + (scol & 15) + ((skl >> 4) << 4)) << 4)
                           + (skl & 15);
      if (USE_WS) {
        // pre-issue chunk0 -> buf1 (halves 1024): waves 0..3 move 512B each
        if (wave < 4) {
          const unsigned l = ldsWt + 2 * W3_H + wave * 512 + lane * 16;
          const unsigned long long gsrc =
              (unsigned long long)(size_t)(wsp + 9 * W2_H) + wave * 512 + lane * 16;
          async_ld_512(l, gsrc);
        }
      }
      for (int kc = 0; kc < 8; ++kc) {
        if (USE_WS) {
          if (kc < 7) {
            if (wave < 4) {
              const unsigned l = ldsWt + (unsigned)((kc & 1) * (2 * W3_H)) + wave * 512 + lane * 16;
              const unsigned long long gsrc =
                  (unsigned long long)(size_t)(wsp + 9 * W2_H + (size_t)(kc + 1) * W3_H)
                  + wave * 512 + lane * 16;
              async_ld_512(l, gsrc);
            }
            WAIT_ASYNC_1();
          } else {
            WAIT_ASYNC_0();
          }
          __syncthreads();
        } else {
          __syncthreads();
          {
            v4h hv;
#pragma unroll
            for (int i = 0; i < 4; ++i) {
              const int krow = kc * 32 + skl + i;
              float v = 0.f;
              if (scol < RAWD)       v = W3p[krow * RAWD + scol];
              else if (scol == RAWD) v = Woccp[krow];
              hv[i] = (_Float16)v;
            }
            *(v4h*)sdst3 = hv;
          }
          __syncthreads();
        }
        const _Float16* W3t = USE_WS ? (Wt + ((kc & 1) ^ 1) * W3_H) : Wt;
        v16h a  = load_a_frag(hw, lane, kc * 32);
        v16h b0 = load_b_frag_swz(W3t, 0, lane);
        acc3[0] = wmma16(a, b0, acc3[0]);
        v16h b1f = load_b_frag_swz(W3t, 1, lane);
        acc3[1] = wmma16(a, b1f, acc3[1]);
        __syncthreads();
      }
    }

    // ---------- per-part epilogue: bias, sigmoid, flag-mask, accumulate ----------
    float flg[8];
#pragma unroll
    for (int r = 0; r < 8; ++r) {
      const int n = wrow0 + r + g * 8;
      flg[r] = (n < N && tflag[(long)n * PPARTS + p] != 0) ? 1.f : 0.f;
    }
    {
      const float bias0 = b3[(long)p * RAWD + ln];
#pragma unroll
      for (int r = 0; r < 8; ++r) accOut0[r] += (acc3[0][r] + bias0) * flg[r];

      const int col = 16 + ln;
      if (col < RAWD) {
        const float bias = b3[(long)p * RAWD + col];
#pragma unroll
        for (int r = 0; r < 8; ++r) accOut1[r] += (acc3[1][r] + bias) * flg[r];
      } else if (col == RAWD) {
        const float bias = bocc[p];
#pragma unroll
        for (int r = 0; r < 8; ++r) {
          const float o = flg[r] / (1.f + __expf(-(acc3[1][r] + bias)));
          accOut1[r] += o;
          occs[(wave * 16 + r + g * 8) * PPARTS + p] = o;
        }
      }
    }
    __syncthreads();
  } // parts

  // ---------- write output (N x 27): [raw(20) | occ(1) | tocc(6)] ----------
  const float inv = 1.f / (float)PPARTS;
#pragma unroll
  for (int r = 0; r < 8; ++r) {
    const int n = wrow0 + r + g * 8;
    if (n < N) out[(long)n * OUTC + ln] = accOut0[r] * inv;
  }
  const int col = 16 + ln;
  if (col <= 20) {
#pragma unroll
    for (int r = 0; r < 8; ++r) {
      const int n = wrow0 + r + g * 8;
      if (n < N) out[(long)n * OUTC + col] = accOut1[r] * inv;
    }
  } else if (col <= 26) {
#pragma unroll
    for (int r = 0; r < 8; ++r) {
      const int n = wrow0 + r + g * 8;
      if (n < N) out[(long)n * OUTC + col] = occs[(wave * 16 + r + g * 8) * PPARTS + (col - 21)];
    }
  }
}

extern "C" void kernel_launch(void* const* d_in, const int* in_sizes, int n_in,
                              void* d_out, int out_size, void* d_ws, size_t ws_size,
                              hipStream_t stream) {
  const float* tpts    = (const float*)d_in[0];
  const float* bigpts  = (const float*)d_in[1];
  const float* viewdir = (const float*)d_in[2];
  // d_in[3] = dists, d_in[4] = part_dist: unused by the reference output
  const float* frame   = (const float*)d_in[5];
  const float* W1      = (const float*)d_in[6];
  const float* b1      = (const float*)d_in[7];
  const float* W2      = (const float*)d_in[8];
  const float* b2      = (const float*)d_in[9];
  const float* W3      = (const float*)d_in[10];
  const float* b3      = (const float*)d_in[11];
  const float* Wocc    = (const float*)d_in[12];
  const float* bocc    = (const float*)d_in[13];
  const int*   tflag   = (const int*)d_in[14];
  float* out = (float*)d_out;

  const int N = in_sizes[0] / (PPARTS * 3);
  const int blocks = (N + 127) / 128;
  const size_t shmem = (size_t)(2 * W2_H * 2)        // double-buffered weight tile
                     + (size_t)(8 * 16 * DDIM * 2)   // per-wave activations
                     + (size_t)(8 * 16 * PPARTS * 4);// occ stash
  const size_t ws_needed = (size_t)PPARTS * PART_H * sizeof(_Float16); // 983,040 B

  if (ws_size >= ws_needed) {
    convert_weights_kernel<<<PPARTS * 10, 256, 0, stream>>>(W1, W2, W3, Wocc, (_Float16*)d_ws);
    tpose_mlp_wmma_kernel<true><<<blocks, 256, shmem, stream>>>(
        tpts, bigpts, viewdir, frame, W1, b1, W2, b2, W3, b3, Wocc, bocc,
        tflag, (const _Float16*)d_ws, out, N);
  } else {
    tpose_mlp_wmma_kernel<false><<<blocks, 256, shmem, stream>>>(
        tpts, bigpts, viewdir, frame, W1, b1, W2, b2, W3, b3, Wocc, bocc,
        tflag, (const _Float16*)nullptr, out, N);
  }
}
